// GroupAttention_46815143526494
// MI455X (gfx1250) — compile-verified, matrix operand order checked by reading, not run on hardware
//
#include <hip/hip_runtime.h>
#include <hip/hip_bf16.h>
#include <math.h>

#define NB 8
#define NS 1024
#define ND 512

typedef __bf16 bf16;
typedef __attribute__((ext_vector_type(16))) __bf16 v16bf;
typedef __attribute__((ext_vector_type(8)))  __bf16 v8bf;
typedef __attribute__((ext_vector_type(8)))  float  v8f;

#define SQRT_1EM9 3.16227766016838e-5f

// ---------------------------------------------------------------------------
// Kernel 1: LayerNorm (eps=1e-6, biased variance) + convert to bf16.
// One block (256 thr) per row of (B*S, D).
// ---------------------------------------------------------------------------
__global__ void ga_ln_bf16(const float* __restrict__ ctx,
                           const float* __restrict__ gamma,
                           const float* __restrict__ beta,
                           unsigned short* __restrict__ xbf_us) {
    const int row = blockIdx.x;
    const float* x = ctx + (size_t)row * ND;
    bf16* xbf = reinterpret_cast<bf16*>(xbf_us) + (size_t)row * ND;

    const int t = threadIdx.x;              // 0..255, each handles 2 elems
    float v0 = x[t], v1 = x[t + 256];

    __shared__ float rs[256], rss[256];
    rs[t]  = v0 + v1;
    rss[t] = v0 * v0 + v1 * v1;
    __syncthreads();
    for (int off = 128; off > 0; off >>= 1) {
        if (t < off) { rs[t] += rs[t + off]; rss[t] += rss[t + off]; }
        __syncthreads();
    }
    const float mu   = rs[0] * (1.0f / ND);
    const float var  = rss[0] * (1.0f / ND) - mu * mu;
    const float rstd = rsqrtf(var + 1e-6f);

    xbf[t]       = (bf16)(((v0 - mu) * rstd) * gamma[t]       + beta[t]);
    xbf[t + 256] = (bf16)(((v1 - mu) * rstd) * gamma[t + 256] + beta[t + 256]);
}

// ---------------------------------------------------------------------------
// Kernel 2: f32 -> bf16 conversion (for Wq / Wk).
// ---------------------------------------------------------------------------
__global__ void ga_cvt_bf16(const float* __restrict__ src,
                            unsigned short* __restrict__ dst_us, int n) {
    int i = blockIdx.x * blockDim.x + threadIdx.x;
    if (i < n) {
        bf16 h = (bf16)src[i];
        reinterpret_cast<bf16*>(dst_us)[i] = h;
    }
}

// ---------------------------------------------------------------------------
// Kernel 3: q = x @ Wq^T + bq ; k = x @ Wk^T + bk via v_wmma_f32_16x16x32_bf16.
// One wave32 per 16x16 output tile. blockDim = 128 (4 waves).
// grid = (M/16, (N/16)/4, 2)  [z: 0 -> q, 1 -> k]
// A = x (row-major MxK).  B(KxN)[d,n] = W[n,d], so the B operand (striped as
// B^T rows across lanes) is just row n of W — contiguous loads.
// ISA 16-bit operand layout: lane m     holds K = {k0..k0+7, k0+16..k0+23},
//                            lane m+16  holds K = {k0+8..k0+15, k0+24..k0+31}.
// ---------------------------------------------------------------------------
__global__ void ga_gemm_qk(const unsigned short* __restrict__ xbf_us,
                           const unsigned short* __restrict__ wq_us,
                           const float* __restrict__ bq,
                           const unsigned short* __restrict__ wk_us,
                           const float* __restrict__ bk,
                           float* __restrict__ qout,
                           float* __restrict__ kout) {
    const bf16* x = reinterpret_cast<const bf16*>(xbf_us);
    const bf16* w = reinterpret_cast<const bf16*>(blockIdx.z ? wk_us : wq_us);
    const float* bias = blockIdx.z ? bk : bq;
    float* out = blockIdx.z ? kout : qout;

    const int wave = threadIdx.x >> 5;
    const int lane = threadIdx.x & 31;
    const int m0 = blockIdx.x * 16;
    const int n0 = (blockIdx.y * 4 + wave) * 16;

    const int mn  = lane & 15;      // row (A) / col (B) within tile
    const int l16 = lane >> 4;      // lane half selects K sub-blocks

    const bf16* xrow = x + (size_t)(m0 + mn) * ND;
    const bf16* wrow = w + (size_t)(n0 + mn) * ND;

    v8f acc = {};
#pragma unroll 4
    for (int k0 = 0; k0 < ND; k0 += 32) {
        const int ko1 = k0 + l16 * 8;
        const int ko2 = k0 + 16 + l16 * 8;
        v8bf alo = *(const v8bf*)(xrow + ko1);
        v8bf ahi = *(const v8bf*)(xrow + ko2);
        v8bf blo = *(const v8bf*)(wrow + ko1);
        v8bf bhi = *(const v8bf*)(wrow + ko2);
        v16bf a, b;
#pragma unroll
        for (int i = 0; i < 8; ++i) {
            a[i] = alo[i]; a[i + 8] = ahi[i];
            b[i] = blo[i]; b[i + 8] = bhi[i];
        }
        acc = __builtin_amdgcn_wmma_f32_16x16x32_bf16(
            false, a, false, b, (short)0, acc, false, false);
    }

    // C/D layout: VGPR v, lanes 0-15 -> (m0+v, n0+lane); lanes 16-31 -> (m0+8+v, ...)
    const int outcol = n0 + mn;
    const float bv = bias[outcol];
    const int rbase = m0 + l16 * 8;
#pragma unroll
    for (int v = 0; v < 8; ++v)
        out[(size_t)(rbase + v) * ND + outcol] = acc[v] + bv;
}

// ---------------------------------------------------------------------------
// Kernel 4: band scores + 2-entry masked softmax + symmetric band weight w +
// L = log(prior-blended super-diag) + exclusive prefix scan C.
// One block of 1024 threads per batch. (Degenerate "entire row masked" case
// — impossible with the all-ones eos_mask — is not reproduced.)
// ---------------------------------------------------------------------------
__global__ void ga_band(const float* __restrict__ q,
                        const float* __restrict__ k,
                        const float* __restrict__ prior,
                        const unsigned char* __restrict__ eos,
                        float* __restrict__ wband,
                        float* __restrict__ Cpre) {
    const int b = blockIdx.x;
    const int s = threadIdx.x;                    // 0..S-1
    const float* qb = q + (size_t)b * NS * ND;
    const float* kb = k + (size_t)b * NS * ND;

    __shared__ float sh_sup[NS];                  // scores[s, s+1]
    __shared__ float sh_sub[NS];                  // scores[s+1, s]
    __shared__ float sh_nsp[NS];                  // neibor[s, s+1]
    __shared__ float sh_nsb[NS];                  // neibor[s, s-1]
    __shared__ float sc[NS];

    float ssup = 0.f, ssub = 0.f;
    if (s < NS - 1) {
        const float4* qs  = (const float4*)(qb + (size_t)s * ND);
        const float4* qs1 = (const float4*)(qb + (size_t)(s + 1) * ND);
        const float4* ks  = (const float4*)(kb + (size_t)s * ND);
        const float4* ks1 = (const float4*)(kb + (size_t)(s + 1) * ND);
        for (int d = 0; d < ND / 4; ++d) {
            float4 a = qs[d],  c = ks1[d];
            float4 e = qs1[d], f = ks[d];
            ssup += a.x * c.x + a.y * c.y + a.z * c.z + a.w * c.w;
            ssub += e.x * f.x + e.y * f.y + e.z * f.z + e.w * f.w;
        }
        ssup *= (1.0f / 256.0f);                  // reference hardcodes 256.0
        ssub *= (1.0f / 256.0f);
    }
    sh_sup[s] = ssup;
    sh_sub[s] = ssub;
    __syncthreads();

    // Row-s softmax over its (<=2) unmasked band entries.
    const size_t rb = (size_t)b * NS * NS + (size_t)s * NS;
    const bool vsup = (s < NS - 1) && eos[rb + s + 1];
    const bool vsub = (s > 0)      && eos[rb + s - 1];
    const float a1 = vsup ? sh_sup[s]     : -1e9f;   // scores[s, s+1]
    const float a2 = vsub ? sh_sub[s - 1] : -1e9f;   // scores[s, s-1]
    const float m  = fmaxf(a1, a2);
    const float e1 = __expf(a1 - m), e2 = __expf(a2 - m);
    const float inv = 1.0f / (e1 + e2);
    sh_nsp[s] = e1 * inv;
    sh_nsb[s] = e2 * inv;
    __syncthreads();

    // Symmetric band weight at (s, s+1): sqrt(n[s,s+1] * n[s+1,s] + 1e-9).
    float wv = 0.f;
    if (s < NS - 1) wv = sqrtf(sh_nsp[s] * sh_nsb[s + 1] + 1e-9f);
    wband[(size_t)b * NS + s] = wv;

    // L[s] = log(prior-blended neibor at super-diag + 1e-9)
    float L = 0.f;
    if (s < NS - 1) {
        const float p = prior[rb + s + 1];
        L = __logf(p + (1.0f - p) * wv + 1e-9f);
    }

    // Hillis-Steele inclusive scan -> exclusive C.
    sc[s] = L;
    __syncthreads();
    for (int off = 1; off < NS; off <<= 1) {
        const float v = (s >= off) ? sc[s - off] : 0.0f;
        __syncthreads();
        sc[s] += v;
        __syncthreads();
    }
    Cpre[(size_t)b * NS + s] = (s == 0) ? 0.0f : sc[s - 1];
}

// ---------------------------------------------------------------------------
// Kernel 5: bandwidth-bound epilogue. One block (256 thr) per output row;
// each thread streams one float4 of the 1024-wide row.
//   neibor[i,j] = p + (1-p) * (|i-j|==1 ? w[min(i,j)] : sqrt(1e-9))
//   g_attn[i,j] = i==j ? neibor[i,i] : exp(C[max]-C[min]) + 1e-9
// ---------------------------------------------------------------------------
__global__ void ga_out(const float* __restrict__ prior,
                       const float* __restrict__ wband,
                       const float* __restrict__ Cpre,
                       float* __restrict__ g_attn,
                       float* __restrict__ neib) {
    const int b = blockIdx.x >> 10;
    const int i = blockIdx.x & (NS - 1);
    const size_t rowoff = (size_t)blockIdx.x * NS;
    const float* Cb = Cpre + (size_t)b * NS;
    const float* Wb = wband + (size_t)b * NS;
    const float Ci = Cb[i];

    const int t = threadIdx.x;                    // 0..255 -> float4 lane
    const int j0 = t * 4;
    float4 p4 = ((const float4*)(prior + rowoff))[t];
    float4 g4, n4;
    const float* pp = &p4.x;
    float* gg = &g4.x;
    float* nn = &n4.x;
#pragma unroll
    for (int u = 0; u < 4; ++u) {
        const int j = j0 + u;
        const int d = j - i;
        const float p = pp[u];
        const float v = (d == 1 || d == -1) ? Wb[(d == 1) ? i : j] : SQRT_1EM9;
        const float no = p + (1.0f - p) * v;
        float g;
        if (d == 0) {
            g = no;
        } else {
            const float Cj = Cb[j];
            const float hi = (d > 0) ? Cj : Ci;
            const float lo = (d > 0) ? Ci : Cj;
            g = __expf(hi - lo) + 1e-9f;
        }
        gg[u] = g;
        nn[u] = no;
    }
    ((float4*)(g_attn + rowoff))[t] = g4;
    ((float4*)(neib + rowoff))[t]   = n4;
}

// ---------------------------------------------------------------------------
extern "C" void kernel_launch(void* const* d_in, const int* in_sizes, int n_in,
                              void* d_out, int out_size, void* d_ws, size_t ws_size,
                              hipStream_t stream) {
    const float*         ctx   = (const float*)d_in[0];
    const unsigned char* eos   = (const unsigned char*)d_in[1];
    const float*         prior = (const float*)d_in[2];
    const float*         gamma = (const float*)d_in[3];
    const float*         beta  = (const float*)d_in[4];
    const float*         Wq    = (const float*)d_in[5];
    const float*         bq    = (const float*)d_in[6];
    const float*         Wk    = (const float*)d_in[7];
    const float*         bk    = (const float*)d_in[8];

    // Workspace layout (bytes):
    char* ws = (char*)d_ws;
    const size_t MiB = 1u << 20;
    unsigned short* xbf  = (unsigned short*)(ws);                     //  8 MiB
    unsigned short* wqb  = (unsigned short*)(ws + 8 * MiB);           // .5 MiB
    unsigned short* wkb  = (unsigned short*)(ws + 8 * MiB + 512 * 1024);
    float* q     = (float*)(ws + 9 * MiB);                            // 16 MiB
    float* k     = (float*)(ws + 25 * MiB);                           // 16 MiB
    float* wband = (float*)(ws + 41 * MiB);                           // 32 KiB
    float* Cpre  = (float*)(ws + 41 * MiB + 32 * 1024);               // 32 KiB

    float* g_attn = (float*)d_out;
    float* neib   = g_attn + (size_t)NB * NS * NS;

    // 1) LayerNorm -> bf16
    ga_ln_bf16<<<dim3(NB * NS), dim3(256), 0, stream>>>(ctx, gamma, beta, xbf);

    // 2) Weights -> bf16
    const int nw = ND * ND;
    ga_cvt_bf16<<<dim3((nw + 255) / 256), dim3(256), 0, stream>>>(Wq, wqb, nw);
    ga_cvt_bf16<<<dim3((nw + 255) / 256), dim3(256), 0, stream>>>(Wk, wkb, nw);

    // 3) q/k projection GEMMs via WMMA bf16 (z: 0 -> q, 1 -> k)
    ga_gemm_qk<<<dim3((NB * NS) / 16, (ND / 16) / 4, 2), dim3(128), 0, stream>>>(
        xbf, wqb, bq, wkb, bk, q, k);

    // 4) band softmax + weights + prefix scan (one block per batch)
    ga_band<<<dim3(NB), dim3(NS), 0, stream>>>(q, k, prior, eos, wband, Cpre);

    // 5) streamed epilogue (dominant, bandwidth-bound)
    ga_out<<<dim3(NB * NS), dim3(256), 0, stream>>>(prior, wband, Cpre, g_attn, neib);
}
